// HomoGCN_50285477102131
// MI455X (gfx1250) — matrix operand order repaired and checked by reading, not compiled, not run on hardware
//
#include <hip/hip_runtime.h>

// ---------------------------------------------------------------------------
// Problem constants (match reference)
// ---------------------------------------------------------------------------
#define N_NODES   100000
#define N_EDGES   1600000
#define IN_DIM    128
#define HID       256
#define N_GRAPHS  256
#define N_OUT     2

// ---------------------------------------------------------------------------
// WMMA vector types (CDNA5 / gfx1250, wave32)
// ---------------------------------------------------------------------------
typedef __attribute__((ext_vector_type(16))) __bf16 v16bf;
typedef __attribute__((ext_vector_type(8)))  float  v8f;

union BF16x16 {
    unsigned short u[16];
    uint4          q[2];
    v16bf          v;
};

// f32 -> bf16 round-to-nearest-even
__device__ __forceinline__ unsigned short f2bf(float f) {
    unsigned u = __builtin_bit_cast(unsigned, f);
    unsigned r = u + 0x7FFFu + ((u >> 16) & 1u);
    return (unsigned short)(r >> 16);
}

// ---------------------------------------------------------------------------
// Bulk f32 -> bf16 row-major convert (8 elements / thread, b128 in, b128 out)
// ---------------------------------------------------------------------------
__global__ __launch_bounds__(256) void cvt_f32_bf16(
    const float* __restrict__ in, unsigned short* __restrict__ out, int n8)
{
    int t = blockIdx.x * 256 + threadIdx.x;
    if (t >= n8) return;
    const float4* p = (const float4*)in + (size_t)t * 2;
    float4 a = p[0], b = p[1];
    union { unsigned short u[8]; uint4 q; } o;
    o.u[0] = f2bf(a.x); o.u[1] = f2bf(a.y); o.u[2] = f2bf(a.z); o.u[3] = f2bf(a.w);
    o.u[4] = f2bf(b.x); o.u[5] = f2bf(b.y); o.u[6] = f2bf(b.z); o.u[7] = f2bf(b.w);
    ((uint4*)out)[t] = o.q;
}

// ---------------------------------------------------------------------------
// Weight swizzle: fp32 row-major W[K,N] -> bf16 in WMMA B-fragment order.
// out[((kt*(N/16)+nt)*32 + lane)*16 + i]
//   = bf16( W[ kt*32 + (lane>>4)*16 + i , nt*16 + (lane&15) ] )
// ---------------------------------------------------------------------------
__global__ __launch_bounds__(256) void swizzle_b(
    const float* __restrict__ W, unsigned short* __restrict__ out, int K, int N)
{
    int t = blockIdx.x * blockDim.x + threadIdx.x;
    if (t >= K * N) return;
    int i    = t & 15;
    int L    = (t >> 4) & 31;
    int rest = t >> 9;
    int ntCount = N >> 4;
    int nt = rest % ntCount;
    int kt = rest / ntCount;
    int k   = (kt << 5) + ((L >> 4) << 4) + i;
    int col = (nt << 4) + (L & 15);
    out[t] = f2bf(W[(size_t)k * N + col]);
}

// ---------------------------------------------------------------------------
// GEMM: C[M,N] = relu( A1[M,K1] @ B1 + (A2[M,K2] @ B2) + bias )
//   A: bf16 row-major; fragment = two aligned b128 loads (no repacking).
//   B: pre-swizzled bf16; within a K-step, tile nt is at constant byte
//      offset nt*1024 from a single loop-carried base pointer, so all tile
//      addresses fold into the b128 instruction offset field.
//   C: written fp32 (Cf) or bf16 row-major (Cb) -- exactly one non-null.
// Each wave computes a 16x128 strip: 1 A-frag feeds 8 WMMAs per K-step.
// All 8 B fragments are loaded into distinct registers before the WMMA
// burst so load latency overlaps the WMMA pipe (no per-op loadcnt 0 waits).
// ---------------------------------------------------------------------------
__global__ __launch_bounds__(256) void gemm_bias_relu(
    const unsigned short* __restrict__ A1, const unsigned short* __restrict__ B1, int K1,
    const unsigned short* __restrict__ A2, const unsigned short* __restrict__ B2, int K2,
    const float* __restrict__ bias, float* __restrict__ Cf,
    unsigned short* __restrict__ Cb, int M, int N)
{
    const int tilesM = M >> 4;
    const int tilesN = N >> 7;                 // 128-wide strips
    const int w = blockIdx.x * (blockDim.x >> 5) + (threadIdx.x >> 5);
    if (w >= tilesM * tilesN) return;          // whole-wave exit: EXEC stays all-ones
    const int tileM = w % tilesM;
    const int tileN = w / tilesM;

    const int lane = threadIdx.x & 31;
    const int half = lane >> 4;
    const int r16  = lane & 15;
    const int row  = (tileM << 4) + r16;
    const int ntStride = N >> 4;               // B K-step advance = ntStride*64 uint4s

    v8f acc[8] = {};

    for (int pass = 0; pass < 2; ++pass) {
        const unsigned short* A = pass ? A2 : A1;
        const unsigned short* B = pass ? B2 : B1;
        const int K = pass ? K2 : K1;
        if (A == nullptr) continue;

        // A: 4 uint4 per 32-K bf16 chunk; fragment = units {2*half, 2*half+1}
        const uint4* ap = (const uint4*)(A + (size_t)row * K);
        // B: base for this strip; tile nt at constant offset nt*64 uint4s
        const uint4* bp = (const uint4*)(B +
            ((((size_t)tileN << 3) * 32 + lane) << 4));

        for (int kt = 0; kt < (K >> 5); ++kt) {
            BF16x16 af;
            af.q[0] = ap[2 * half];
            af.q[1] = ap[2 * half + 1];

            BF16x16 bfrag[8];
#pragma unroll
            for (int nt = 0; nt < 8; ++nt) {
                bfrag[nt].q[0] = bp[nt * 64 + 0];
                bfrag[nt].q[1] = bp[nt * 64 + 1];
            }
#pragma unroll
            for (int nt = 0; nt < 8; ++nt) {
                acc[nt] = __builtin_amdgcn_wmma_f32_16x16x32_bf16(
                    false, af.v, false, bfrag[nt].v, (short)0, acc[nt], false, false);
            }
            ap += 4;                 //  64 B  (32 bf16 K values)
            bp += ntStride * 64;     //  K-step advance in B
        }
    }

    // epilogue: C/D layout -> row = v + 8*half, col = lane&15 within tile
    if (Cf) {
#pragma unroll
        for (int nt = 0; nt < 8; ++nt) {
            const int col = (((tileN << 3) + nt) << 4) + r16;
            const float bv = bias[col];
#pragma unroll
            for (int v = 0; v < 8; ++v) {
                const int r = (tileM << 4) + v + (half << 3);
                float val = acc[nt][v] + bv;
                Cf[(size_t)r * N + col] = val > 0.f ? val : 0.f;
            }
        }
    } else {
#pragma unroll
        for (int nt = 0; nt < 8; ++nt) {
            const int col = (((tileN << 3) + nt) << 4) + r16;
            const float bv = bias[col];
#pragma unroll
            for (int v = 0; v < 8; ++v) {
                const int r = (tileM << 4) + v + (half << 3);
                float val = acc[nt][v] + bv;
                Cb[(size_t)r * N + col] = f2bf(val > 0.f ? val : 0.f);
            }
        }
    }
}

// ---------------------------------------------------------------------------
// Edge scatter: neigh[dst] = max(neigh[dst], m[src]) per feature.
// m >= 0 (post-ReLU) -> float bits monotone as u32 -> atomicMax(u32) with
// zero-init. Skip exact-zero values (~half, post-ReLU): they can never beat
// the zero-initialized buffer.
// One wave per edge: 32 lanes x 4 floats = 128 features.
// ---------------------------------------------------------------------------
__global__ __launch_bounds__(256) void edge_scatter_max(
    const unsigned* __restrict__ m, const int* __restrict__ src,
    const int* __restrict__ dst, unsigned* __restrict__ neigh, int nE)
{
    int e = blockIdx.x * 8 + (threadIdx.x >> 5);
    if (e >= nE) return;
    int lane = threadIdx.x & 31;
    int s = src[e];
    int d = dst[e];
    uint4 v = ((const uint4*)(m + (size_t)s * IN_DIM))[lane];
    unsigned* o = neigh + (size_t)d * IN_DIM + (lane << 2);
    if (v.x) atomicMax(o + 0, v.x);
    if (v.y) atomicMax(o + 1, v.y);
    if (v.z) atomicMax(o + 2, v.z);
    if (v.w) atomicMax(o + 3, v.w);
}

// ---------------------------------------------------------------------------
// Per-(graph,feature) max of h. graph_id is SORTED, so each block privately
// reduces 32 consecutive nodes per feature and flushes one atomic per
// segment change (~32x fewer atomics). h >= 0 -> u32 atomicMax trick.
// Grid: N_NODES/32 blocks x 256 threads (thread = feature).
// ---------------------------------------------------------------------------
__global__ __launch_bounds__(256) void graph_feat_max(
    const float* __restrict__ h, const int* __restrict__ gid,
    unsigned* __restrict__ gmax)
{
    int f = threadIdx.x;
    int base = blockIdx.x * 32;
    int curG = gid[base];
    float curM = 0.f;                        // identity for non-negative h
    for (int i = 0; i < 32; ++i) {
        int node = base + i;
        int g = gid[node];
        float v = h[((size_t)node << 8) + f];
        if (g != curG) {
            atomicMax(gmax + (((size_t)curG) << 8) + f,
                      __builtin_bit_cast(unsigned, curM));
            curG = g;
            curM = v;
        } else {
            curM = fmaxf(curM, v);
        }
    }
    atomicMax(gmax + (((size_t)curG) << 8) + f, __builtin_bit_cast(unsigned, curM));
}

// ---------------------------------------------------------------------------
// zsum[g][f] += exp(h - gmax[g][f]), same sorted-segment blocking.
// ---------------------------------------------------------------------------
__global__ __launch_bounds__(256) void graph_expsum(
    const float* __restrict__ h, const int* __restrict__ gid,
    const float* __restrict__ gmax, float* __restrict__ zsum)
{
    int f = threadIdx.x;
    int base = blockIdx.x * 32;
    int curG = gid[base];
    float gm = gmax[(((size_t)curG) << 8) + f];
    float curS = 0.f;
    for (int i = 0; i < 32; ++i) {
        int node = base + i;
        int g = gid[node];
        float v = h[((size_t)node << 8) + f];
        if (g != curG) {
            atomicAdd(zsum + (((size_t)curG) << 8) + f, curS);
            curG = g;
            gm = gmax[(((size_t)g) << 8) + f];
            curS = __expf(v - gm);
        } else {
            curS += __expf(v - gm);
        }
    }
    atomicAdd(zsum + (((size_t)curG) << 8) + f, curS);
}

// ---------------------------------------------------------------------------
// softmax_nodes + max_nodes collapses to 1/zsum:
//   max_n exp(h-gmax)/zsum = exp(0)/zsum = 1/zsum  (0 for empty graphs)
// ---------------------------------------------------------------------------
__global__ __launch_bounds__(256) void finalize_inv(
    const float* __restrict__ zsum, float* __restrict__ finalM)
{
    int i = blockIdx.x * 256 + threadIdx.x;   // 256*256 = 65536
    float s = zsum[i];
    finalM[i] = s > 0.f ? 1.0f / s : 0.f;
}

// ---------------------------------------------------------------------------
// Readout: out[G, 2] = final[G, 256] @ Wr[256, 2] + br (tiny -> plain VALU)
// ---------------------------------------------------------------------------
__global__ __launch_bounds__(256) void readout(
    const float* __restrict__ finalM, const float* __restrict__ Wr,
    const float* __restrict__ br, float* __restrict__ out)
{
    int t = blockIdx.x * 256 + threadIdx.x;   // 512 threads
    if (t >= N_GRAPHS * N_OUT) return;
    int g = t >> 1;
    int o = t & 1;
    float s = br[o];
    const float* fr = finalM + ((size_t)g << 8);
#pragma unroll 8
    for (int k = 0; k < HID; ++k) s += fr[k] * Wr[k * N_OUT + o];
    out[t] = s;
}

// ---------------------------------------------------------------------------
// Host-side orchestration
// ---------------------------------------------------------------------------
extern "C" void kernel_launch(void* const* d_in, const int* in_sizes, int n_in,
                              void* d_out, int out_size, void* d_ws, size_t ws_size,
                              hipStream_t stream)
{
    (void)in_sizes; (void)n_in; (void)out_size; (void)ws_size;

    const float* x       = (const float*)d_in[0];
    const float* W_pool  = (const float*)d_in[1];
    const float* b_pool  = (const float*)d_in[2];
    const float* W_self  = (const float*)d_in[3];
    const float* W_neigh = (const float*)d_in[4];
    const float* b_sage  = (const float*)d_in[5];
    const float* W1      = (const float*)d_in[6];
    const float* b1      = (const float*)d_in[7];
    const float* W2      = (const float*)d_in[8];
    const float* b2      = (const float*)d_in[9];
    const float* Wr      = (const float*)d_in[10];
    const float* br      = (const float*)d_in[11];
    const int*   src     = (const int*)d_in[12];
    const int*   dst     = (const int*)d_in[13];
    const int*   gid     = (const int*)d_in[14];
    float* out = (float*)d_out;
    char*  ws  = (char*)d_ws;

    // ---- workspace layout (all offsets 256B-aligned) -----------------------
    const size_t SZ_M   = (size_t)N_NODES * IN_DIM * 4;   //  51,200,000
    const size_t SZ_XBF = (size_t)N_NODES * IN_DIM * 2;   //  25,600,000
    const size_t SZ_HBF = (size_t)N_NODES * HID * 2;      //  51,200,000

    const size_t OFF_M     = 0;                    // m fp32; later h3 fp32 region
    const size_t OFF_NEIGH = OFF_M + SZ_M;         // neigh fp32 (atomics)
    const size_t OFF_XBF   = OFF_NEIGH + SZ_M;     // x bf16
    const size_t OFF_NBF   = OFF_XBF + SZ_XBF;     // neigh bf16
    const size_t OFF_H1BF  = OFF_NBF + SZ_XBF;     // h1 bf16
    const size_t OFF_H2BF  = OFF_XBF;              // reuse xbf+nbf after GEMM2
    const size_t OFF_H3    = OFF_M;                // reuse m+neigh after GEMM3
    const size_t OFF_WSW   = OFF_H1BF + SZ_HBF;
    const size_t OFF_WPOOL  = OFF_WSW;
    const size_t OFF_WSELF  = OFF_WPOOL  + (size_t)IN_DIM * IN_DIM * 2;
    const size_t OFF_WNEIGH = OFF_WSELF  + (size_t)IN_DIM * HID * 2;
    const size_t OFF_W1     = OFF_WNEIGH + (size_t)IN_DIM * HID * 2;
    const size_t OFF_W2     = OFF_W1     + (size_t)HID * HID * 2;
    const size_t OFF_GMAX   = OFF_W2     + (size_t)HID * HID * 2;
    const size_t OFF_ZSUM   = OFF_GMAX   + (size_t)N_GRAPHS * HID * 4;
    const size_t OFF_FINAL  = OFF_ZSUM   + (size_t)N_GRAPHS * HID * 4;

    float*    m       = (float*)(ws + OFF_M);
    unsigned* mU      = (unsigned*)(ws + OFF_M);
    unsigned* neighU  = (unsigned*)(ws + OFF_NEIGH);
    float*    neighF  = (float*)(ws + OFF_NEIGH);
    unsigned short* xbf   = (unsigned short*)(ws + OFF_XBF);
    unsigned short* nbf   = (unsigned short*)(ws + OFF_NBF);
    unsigned short* h1bf  = (unsigned short*)(ws + OFF_H1BF);
    unsigned short* h2bf  = (unsigned short*)(ws + OFF_H2BF);
    float*    h3      = (float*)(ws + OFF_H3);
    unsigned short* wpool_sw  = (unsigned short*)(ws + OFF_WPOOL);
    unsigned short* wself_sw  = (unsigned short*)(ws + OFF_WSELF);
    unsigned short* wneigh_sw = (unsigned short*)(ws + OFF_WNEIGH);
    unsigned short* w1_sw     = (unsigned short*)(ws + OFF_W1);
    unsigned short* w2_sw     = (unsigned short*)(ws + OFF_W2);
    unsigned* gmaxU  = (unsigned*)(ws + OFF_GMAX);
    float*    gmaxF  = (float*)(ws + OFF_GMAX);
    float*    zsum   = (float*)(ws + OFF_ZSUM);
    float*    finalM = (float*)(ws + OFF_FINAL);

    // ---- zero atomic-accumulation regions (deterministic each launch) ------
    hipMemsetAsync(neighU, 0, SZ_M, stream);
    hipMemsetAsync(gmaxU,  0, (size_t)N_GRAPHS * HID * 4, stream);
    hipMemsetAsync(zsum,   0, (size_t)N_GRAPHS * HID * 4, stream);

    // ---- one-time conversions / swizzles -----------------------------------
    cvt_f32_bf16<<<(N_NODES * IN_DIM / 8 + 255) / 256, 256, 0, stream>>>(
        x, xbf, N_NODES * IN_DIM / 8);
    swizzle_b<<<(IN_DIM * IN_DIM + 255) / 256, 256, 0, stream>>>(W_pool,  wpool_sw,  IN_DIM, IN_DIM);
    swizzle_b<<<(IN_DIM * HID    + 255) / 256, 256, 0, stream>>>(W_self,  wself_sw,  IN_DIM, HID);
    swizzle_b<<<(IN_DIM * HID    + 255) / 256, 256, 0, stream>>>(W_neigh, wneigh_sw, IN_DIM, HID);
    swizzle_b<<<(HID * HID       + 255) / 256, 256, 0, stream>>>(W1,      w1_sw,     HID, HID);
    swizzle_b<<<(HID * HID       + 255) / 256, 256, 0, stream>>>(W2,      w2_sw,     HID, HID);

    const int tilesM = N_NODES / 16;   // 6250

    // ---- GEMM1: m = relu(x @ W_pool + b_pool)   fp32 out (for atomic gather)
    {
        int waves = tilesM * (IN_DIM / 128);
        gemm_bias_relu<<<(waves + 7) / 8, 256, 0, stream>>>(
            xbf, wpool_sw, IN_DIM, nullptr, nullptr, 0,
            b_pool, m, nullptr, N_NODES, IN_DIM);
    }

    // ---- edge gather + segment max (L2-resident random gather) -------------
    edge_scatter_max<<<N_EDGES / 8, 256, 0, stream>>>(mU, src, dst, neighU, N_EDGES);
    cvt_f32_bf16<<<(N_NODES * IN_DIM / 8 + 255) / 256, 256, 0, stream>>>(
        neighF, nbf, N_NODES * IN_DIM / 8);

    // ---- GEMM2: h1 = relu(x @ W_self + neigh @ W_neigh + b_sage)  bf16 out
    {
        int waves = tilesM * (HID / 128);
        gemm_bias_relu<<<(waves + 7) / 8, 256, 0, stream>>>(
            xbf, wself_sw, IN_DIM, nbf, wneigh_sw, IN_DIM,
            b_sage, nullptr, h1bf, N_NODES, HID);
    }

    // ---- GEMM3: h2 = relu(h1 @ W1 + b1)   bf16 out
    {
        int waves = tilesM * (HID / 128);
        gemm_bias_relu<<<(waves + 7) / 8, 256, 0, stream>>>(
            h1bf, w1_sw, HID, nullptr, nullptr, 0,
            b1, nullptr, h2bf, N_NODES, HID);
    }

    // ---- GEMM4: h3 = relu(h2 @ W2 + b2)   fp32 out (for softmax passes)
    {
        int waves = tilesM * (HID / 128);
        gemm_bias_relu<<<(waves + 7) / 8, 256, 0, stream>>>(
            h2bf, w2_sw, HID, nullptr, nullptr, 0,
            b2, h3, nullptr, N_NODES, HID);
    }

    // ---- per-graph softmax + max readout (collapsed to 1/zsum) -------------
    graph_feat_max<<<N_NODES / 32, 256, 0, stream>>>(h3, gid, gmaxU);
    graph_expsum<<<N_NODES / 32, 256, 0, stream>>>(h3, gid, gmaxF, zsum);
    finalize_inv<<<(N_GRAPHS * HID) / 256, 256, 0, stream>>>(zsum, finalM);
    readout<<<2, 256, 0, stream>>>(finalM, Wr, br, out);
}
